// PyramidalAttention_1726576856574
// MI455X (gfx1250) — compile-verified, matrix operand order checked by reading, not run on hardware
//
#include <hip/hip_runtime.h>
#include <hip/hip_bf16.h>
#include <stdint.h>

// ---------------------------------------------------------------------------
// Pyramidal attention for MI455X (gfx1250, wave32, WMMA).
// B=4, S=2720, D=512, H=8, DK=64, M=16.  N = B*S = 10880.
// Pipeline:
//   1) convert weights f32 -> bf16 (once per launch)
//   2) fused LayerNorm -> bf16 q-input (x 1/sqrt(DK)) + bf16 copy of x
//   3) 3x WMMA GEMM (q,k,v) with double-buffered ASYNC global->LDS DMA
//   4) gathered 16-neighbor attention per head (wave per head, shfl reduce)
//   5) WMMA GEMM FC with fused bias + residual epilogue -> d_out (f32)
// ---------------------------------------------------------------------------

typedef __bf16        v16bf __attribute__((ext_vector_type(16)));
typedef float         v8f   __attribute__((ext_vector_type(8)));
typedef unsigned int  u32x4 __attribute__((ext_vector_type(4)));

#define BBATCH 4
#define SSEQ   2720
#define NROW   (BBATCH * SSEQ)   // 10880, divisible by 128
#define DDIM   512
#define HHEAD  8
#define DKDIM  64
#define MNBR   16

#define BM  128
#define BN  128
#define BKK 32
#define NKS (DDIM / BKK)   // 16 k-steps
#define LDT 40             // padded LDS stride in halves (80B rows)

__device__ __forceinline__ unsigned short f2bf(float f) {
  unsigned int u = __float_as_uint(f);
  u += 0x7FFFu + ((u >> 16) & 1u);          // round-to-nearest-even
  return (unsigned short)(u >> 16);
}

// ------------------------------ f32 -> bf16 --------------------------------
__global__ void cvt_f32_bf16(const float* __restrict__ in,
                             unsigned short* __restrict__ out, int n) {
  int i = blockIdx.x * blockDim.x + threadIdx.x;
  if (i < n) out[i] = f2bf(in[i]);
}

// --------------------- LayerNorm + bf16 staging ----------------------------
__global__ __launch_bounds__(256)
void ln_prep(const float* __restrict__ x,
             const float* __restrict__ gamma,
             const float* __restrict__ beta,
             unsigned short* __restrict__ x_bf,
             unsigned short* __restrict__ qin_bf) {
  __shared__ float s1[256];
  __shared__ float s2[256];
  const int tid = threadIdx.x;
  const size_t base = (size_t)blockIdx.x * DDIM;

  float a0 = x[base + tid];
  float a1 = x[base + tid + 256];
  s1[tid] = a0 + a1;
  s2[tid] = a0 * a0 + a1 * a1;
  __syncthreads();
  for (int o = 128; o > 0; o >>= 1) {
    if (tid < o) { s1[tid] += s1[tid + o]; s2[tid] += s2[tid + o]; }
    __syncthreads();
  }
  const float mu   = s1[0] * (1.0f / DDIM);
  const float var  = s2[0] * (1.0f / DDIM) - mu * mu;
  const float rstd = rsqrtf(var + 1e-6f);

  const float g0 = gamma[tid],       b0 = beta[tid];
  const float g1 = gamma[tid + 256], b1 = beta[tid + 256];

  x_bf[base + tid]         = f2bf(a0);
  x_bf[base + tid + 256]   = f2bf(a1);
  // fold q scale 1/sqrt(DK) = 0.125 into the LN output used only by q
  qin_bf[base + tid]       = f2bf(((a0 - mu) * rstd * g0 + b0) * 0.125f);
  qin_bf[base + tid + 256] = f2bf(((a1 - mu) * rstd * g1 + b1) * 0.125f);
}

// --------------------------- WMMA bf16 GEMM --------------------------------
// C[n][o] = sum_k A[n][k] * W[o][k]  (+bias[o] +resid[n][o])
// Double-buffered LDS tiles filled by GLOBAL_LOAD_ASYNC_TO_LDS_B128 (ASYNCcnt).
union FragBF { v16bf v; u32x4 u[2]; };

__global__ __launch_bounds__(256)
void gemm_bf16_wmma(const unsigned short* __restrict__ A,
                    const unsigned short* __restrict__ W,
                    float* __restrict__ C,
                    const float* __restrict__ bias,
                    const float* __restrict__ resid) {
  __shared__ unsigned short lA[2][BM * LDT];
  __shared__ unsigned short lB[2][BN * LDT];

  const int rowBase = blockIdx.x * BM;
  const int colBase = blockIdx.y * BN;
  const int tid  = threadIdx.x;
  const int lane = tid & 31;
  const int wv   = tid >> 5;     // wave 0..7
  const int wm   = wv & 3;       // 4 waves along M: 32 rows each
  const int wn   = wv >> 2;      // 2 waves along N: 64 cols each

  const int ldRow = tid >> 1;          // 0..127: one tile row per 2 threads
  const int ldSeg = (tid & 1) * 16;    // halves: 2x 32B segments per row

  const int fRow = lane & 15;
  const int fKlo = (lane >> 4) << 3;   // lane<16 -> K0=0 ; lane>=16 -> K0=8

  const unsigned short* gA0 = A + (size_t)(rowBase + ldRow) * DDIM + ldSeg;
  const unsigned short* gB0 = W + (size_t)(colBase + ldRow) * DDIM + ldSeg;

  // issue one 32B A-segment + one 32B B-segment as async DMA to LDS.
  // INST_OFFSET applies to both the LDS and global addresses, so the two
  // 16B halves of each segment share one address pair.
  auto issue_tile = [&](int kofs, int buf) {
    unsigned la = (unsigned)(size_t)&lA[buf][ldRow * LDT + ldSeg];
    unsigned lb = (unsigned)(size_t)&lB[buf][ldRow * LDT + ldSeg];
    const unsigned short* pa = gA0 + kofs;
    const unsigned short* pb = gB0 + kofs;
    asm volatile(
        "global_load_async_to_lds_b128 %0, %2, off\n\t"
        "global_load_async_to_lds_b128 %0, %2, off offset:16\n\t"
        "global_load_async_to_lds_b128 %1, %3, off\n\t"
        "global_load_async_to_lds_b128 %1, %3, off offset:16"
        :: "v"(la), "v"(lb), "v"(pa), "v"(pb)
        : "memory");
  };

  v8f acc[2][4] = {};

  issue_tile(0, 0);                          // prologue: tile 0 -> buf 0
  for (int ks = 0; ks < NKS; ++ks) {
    if (ks + 1 < NKS) {
      issue_tile((ks + 1) * BKK, (ks + 1) & 1);   // prefetch next tile
      asm volatile("s_wait_asynccnt 0x4" ::: "memory");  // tile ks landed
    } else {
      asm volatile("s_wait_asynccnt 0x0" ::: "memory");
    }
    __syncthreads();                         // all waves' DMA visible

    const unsigned short* tA = lA[ks & 1];
    const unsigned short* tB = lB[ks & 1];

    // fragments per ISA 16-bit A layout:
    // VGPR0-3 = K[fKlo..fKlo+7], VGPR4-7 = K[fKlo+16..fKlo+23]
    FragBF fa[2], fb[4];
#pragma unroll
    for (int mt = 0; mt < 2; ++mt) {
      const unsigned short* p = &tA[(wm * 32 + mt * 16 + fRow) * LDT + fKlo];
      fa[mt].u[0] = *(const u32x4*)p;
      fa[mt].u[1] = *(const u32x4*)(p + 16);
    }
#pragma unroll
    for (int nt = 0; nt < 4; ++nt) {
      const unsigned short* p = &tB[(wn * 64 + nt * 16 + fRow) * LDT + fKlo];
      fb[nt].u[0] = *(const u32x4*)p;
      fb[nt].u[1] = *(const u32x4*)(p + 16);
    }

#pragma unroll
    for (int mt = 0; mt < 2; ++mt)
#pragma unroll
      for (int nt = 0; nt < 4; ++nt)
        acc[mt][nt] = __builtin_amdgcn_wmma_f32_16x16x32_bf16(
            false, fa[mt].v, false, fb[nt].v, (short)0, acc[mt][nt],
            false, false);

    __syncthreads();                         // buffer free for tile ks+2
  }

  // ---- epilogue: C layout = VGPR j -> row (j | j+8), lane&15 -> col ----
  const int rOff = (lane >> 4) << 3;
  const int cIn  = lane & 15;
#pragma unroll
  for (int mt = 0; mt < 2; ++mt) {
#pragma unroll
    for (int nt = 0; nt < 4; ++nt) {
      const int c  = colBase + wn * 64 + nt * 16 + cIn;
      const int r0 = rowBase + wm * 32 + mt * 16 + rOff;
      const float bv = bias ? bias[c] : 0.0f;
#pragma unroll
      for (int j = 0; j < 8; ++j) {
        const size_t off = (size_t)(r0 + j) * DDIM + c;
        float val = acc[mt][nt][j] + bv;
        if (resid) val += resid[off];
        C[off] = val;
      }
    }
  }
}

// ------------------ gathered 16-neighbor attention -------------------------
__global__ __launch_bounds__(256)
void sparse_attn(const float* __restrict__ q,
                 const float* __restrict__ k,
                 const float* __restrict__ v,
                 const int* __restrict__ qk_mask,
                 unsigned short* __restrict__ attn_bf) {
  const int bs   = blockIdx.x;
  const int s    = bs % SSEQ;
  const int b    = bs / SSEQ;
  const int lane = threadIdx.x & 31;
  const int h    = threadIdx.x >> 5;

  const size_t qbase = (size_t)bs * DDIM + h * DKDIM;
  const float q0 = q[qbase + lane];
  const float q1 = q[qbase + lane + 32];

  int   idxm[MNBR];
  float sc[MNBR];
#pragma unroll
  for (int m = 0; m < MNBR; ++m) {
    const int idx = qk_mask[s * MNBR + m];
    idxm[m] = idx;
    const int row = (idx < 0) ? 0 : idx;
    const size_t kb = ((size_t)b * SSEQ + row) * DDIM + h * DKDIM;
    float d = q0 * k[kb + lane] + q1 * k[kb + lane + 32];
#pragma unroll
    for (int off = 16; off > 0; off >>= 1)
      d += __shfl_xor(d, off, 32);
    sc[m] = (idx < 0) ? 0.0f : d;      // masked scores participate as 0
  }

  float mx = sc[0];
#pragma unroll
  for (int m = 1; m < MNBR; ++m) mx = fmaxf(mx, sc[m]);
  float w[MNBR];
  float sum = 0.0f;
#pragma unroll
  for (int m = 0; m < MNBR; ++m) { w[m] = __expf(sc[m] - mx); sum += w[m]; }
  const float inv = 1.0f / sum;        // no renorm after re-masking (ref-exact)

  float a0 = 0.0f, a1 = 0.0f;
#pragma unroll
  for (int m = 0; m < MNBR; ++m) {
    if (idxm[m] < 0) continue;
    const float wm_ = w[m] * inv;
    const size_t vb = ((size_t)b * SSEQ + idxm[m]) * DDIM + h * DKDIM;
    a0 += wm_ * v[vb + lane];
    a1 += wm_ * v[vb + lane + 32];
  }
  attn_bf[qbase + lane]      = f2bf(a0);
  attn_bf[qbase + lane + 32] = f2bf(a1);
}

// ---------------------------------------------------------------------------
extern "C" void kernel_launch(void* const* d_in, const int* in_sizes, int n_in,
                              void* d_out, int out_size, void* d_ws, size_t ws_size,
                              hipStream_t stream) {
  (void)in_sizes; (void)n_in; (void)out_size; (void)ws_size;

  const float* hidden = (const float*)d_in[0];
  const float* w_q    = (const float*)d_in[1];
  const float* w_k    = (const float*)d_in[2];
  const float* w_v    = (const float*)d_in[3];
  const float* w_fc   = (const float*)d_in[4];
  const float* b_fc   = (const float*)d_in[5];
  const float* ln_g   = (const float*)d_in[6];
  const float* ln_b   = (const float*)d_in[7];
  const int*   qk     = (const int*)d_in[8];
  // d_in[9] (k_q_mask) is unused by the reference

  // ---- workspace layout ----
  unsigned short* x_bf   = (unsigned short*)d_ws;
  unsigned short* qin_bf = x_bf   + (size_t)NROW * DDIM;
  unsigned short* at_bf  = qin_bf + (size_t)NROW * DDIM;
  unsigned short* wq_bf  = at_bf  + (size_t)NROW * DDIM;
  unsigned short* wk_bf  = wq_bf  + (size_t)DDIM * DDIM;
  unsigned short* wv_bf  = wk_bf  + (size_t)DDIM * DDIM;
  unsigned short* wfc_bf = wv_bf  + (size_t)DDIM * DDIM;
  float* qf = (float*)(wfc_bf + (size_t)DDIM * DDIM);
  float* kf = qf + (size_t)NROW * DDIM;
  float* vf = kf + (size_t)NROW * DDIM;

  const int wn = DDIM * DDIM;
  cvt_f32_bf16<<<(wn + 255) / 256, 256, 0, stream>>>(w_q,  wq_bf,  wn);
  cvt_f32_bf16<<<(wn + 255) / 256, 256, 0, stream>>>(w_k,  wk_bf,  wn);
  cvt_f32_bf16<<<(wn + 255) / 256, 256, 0, stream>>>(w_v,  wv_bf,  wn);
  cvt_f32_bf16<<<(wn + 255) / 256, 256, 0, stream>>>(w_fc, wfc_bf, wn);

  ln_prep<<<NROW, 256, 0, stream>>>(hidden, ln_g, ln_b, x_bf, qin_bf);

  dim3 gg(NROW / BM, DDIM / BN);   // 85 x 4
  gemm_bf16_wmma<<<gg, 256, 0, stream>>>(qin_bf, wq_bf, qf, nullptr, nullptr);
  gemm_bf16_wmma<<<gg, 256, 0, stream>>>(x_bf,   wk_bf, kf, nullptr, nullptr);
  gemm_bf16_wmma<<<gg, 256, 0, stream>>>(x_bf,   wv_bf, vf, nullptr, nullptr);

  sparse_attn<<<NROW, 256, 0, stream>>>(qf, kf, vf, qk, at_bf);

  gemm_bf16_wmma<<<gg, 256, 0, stream>>>(at_bf, wfc_bf, (float*)d_out, b_fc, hidden);
}
